// FujiEngramModule_71159018160273
// MI455X (gfx1250) — matrix operand order repaired and checked by reading, not compiled, not run on hardware
//
#include <hip/hip_runtime.h>
#include <hip/hip_bf16.h>

// ---------------------------------------------------------------------------
// Problem constants (from reference)
// ---------------------------------------------------------------------------
#define BB 4
#define SS 4096
#define HSIZE 2048
#define EE 32
#define KSZ 4
#define NHEAD 4
#define NORD 2
#define THEADS 8               // NORD * NHEAD
#define PRIME 500009ull
#define NTOK (BB * SS)         // 16384

typedef __attribute__((ext_vector_type(16))) __bf16 v16bf;
typedef __attribute__((ext_vector_type(8)))  __bf16 v8bf;
typedef __attribute__((ext_vector_type(8)))  float  v8f;
typedef __attribute__((ext_vector_type(4)))  float  f32x4;

// sigmoid via hardware transcendentals: v_exp_f32 + v_add + v_rcp_f32
__device__ __forceinline__ float fast_sigmoid(float x) {
    return __builtin_amdgcn_rcpf(1.f + __expf(-x));
}

// ---------------------------------------------------------------------------
// K0: f32 -> bf16 weight conversion (Wg: 32x2048, Wo: 2048x32)
// ---------------------------------------------------------------------------
__global__ void cvt_bf16_kernel(const float* __restrict__ src,
                                __bf16* __restrict__ dst, int n) {
    int i = blockIdx.x * blockDim.x + threadIdx.x;
    if (i < n) dst[i] = (__bf16)src[i];
}

// ---------------------------------------------------------------------------
// K1: hash + table gather + 256->32 projection.
// One wave (32 lanes) per token; 8 waves per 256-thread block.
// Table gathers use default (RT) caching: they are the L2-hot working set.
// ---------------------------------------------------------------------------
__global__ void hash_gather_proj_kernel(const long long* __restrict__ ids,
                                        const long long* __restrict__ mapping,
                                        const long long* __restrict__ mult,
                                        const float* __restrict__ table,
                                        const float* __restrict__ Wh,
                                        float* __restrict__ emb_pre) {
    __shared__ float     a_sh[8][THEADS * EE];
    __shared__ long long row_sh[8][THEADS];

    const int lane = threadIdx.x & 31;
    const int wv   = threadIdx.x >> 5;           // 0..7
    const int t    = blockIdx.x * 8 + wv;        // token id 0..16383
    const int b    = t >> 12;                    // /SS
    const int s    = t & (SS - 1);

    if (lane < THEADS) {
        const int oi   = lane >> 2;              // 0 -> 2gram, 1 -> 3gram
        const int head = lane & 3;
        const int k    = 2 + oi;
        unsigned long long h = 0ull;
        for (int j = 0; j < k; ++j) {
            int pos = s + j; if (pos > SS - 1) pos = SS - 1;
            long long tok  = ids[(size_t)b * SS + pos];
            long long comp = mapping[tok];
            unsigned long long prod =
                (unsigned long long)comp *
                (unsigned long long)mult[(oi * NHEAD + head) * 3 + j];
            h = (j == 0) ? prod : (h ^ prod);
        }
        row_sh[wv][lane] = (long long)(h % PRIME) +
                           (long long)(oi * NHEAD + head) * (long long)PRIME;
    }
    __syncthreads();

    for (int hh = 0; hh < THEADS; ++hh) {
        long long row = row_sh[wv][hh];
        a_sh[wv][hh * EE + lane] = table[(size_t)row * EE + lane];
    }
    __syncthreads();

    const float* wrow = Wh + (size_t)lane * (THEADS * EE);
    float acc = 0.f;
#pragma unroll 8
    for (int kk = 0; kk < THEADS * EE; ++kk)
        acc += a_sh[wv][kk] * wrow[kk];
    emb_pre[(size_t)t * EE + lane] = acc;
}

// ---------------------------------------------------------------------------
// Helper: pack 4x f32x4 into a bf16 A fragment (contiguous-K 16x32 A layout: k = 16*h + s).
// ---------------------------------------------------------------------------
__device__ __forceinline__ v16bf pack_a(const f32x4 f0, const f32x4 f1,
                                        const f32x4 f2, const f32x4 f3) {
    v16bf a;
    a[0]  = (__bf16)f0.x; a[1]  = (__bf16)f0.y; a[2]  = (__bf16)f0.z; a[3]  = (__bf16)f0.w;
    a[4]  = (__bf16)f1.x; a[5]  = (__bf16)f1.y; a[6]  = (__bf16)f1.z; a[7]  = (__bf16)f1.w;
    a[8]  = (__bf16)f2.x; a[9]  = (__bf16)f2.y; a[10] = (__bf16)f2.z; a[11] = (__bf16)f2.w;
    a[12] = (__bf16)f3.x; a[13] = (__bf16)f3.y; a[14] = (__bf16)f3.z; a[15] = (__bf16)f3.w;
    return a;
}

// ---------------------------------------------------------------------------
// K2: gate = sigmoid(hidden @ Wg^T).
// One wave per 32-token tile (2 M-tiles) -> 4 independent WMMAs per K-step,
// which fills the bf16 WMMA hazard window with useful work.
// hidden is streamed once -> non-temporal loads keep table rows in L2.
// ---------------------------------------------------------------------------
__global__ void gate_wmma_kernel(const float* __restrict__ X,
                                 const __bf16* __restrict__ Wgb,
                                 float* __restrict__ gate) {
    const int lane = threadIdx.x;                // block = exactly 1 wave
    const int row0 = blockIdx.x * 32;

    const int mrow = lane & 15;
    const int klo  = (lane < 16) ? 0 : 16;       // A: VGPR0-3 K group (contiguous K run)
    const int khi  = klo + 8;                    // A: VGPR4-7 K group
    const int kB   = (lane < 16) ? 0 : 16;       // B: contiguous K run
    const int nb   = lane & 15;

    const float* xr0 = X + (size_t)(row0 + mrow) * HSIZE;
    const float* xr1 = X + (size_t)(row0 + 16 + mrow) * HSIZE;

    v8f c00 = {}, c01 = {}, c10 = {}, c11 = {};
    for (int kb = 0; kb < HSIZE; kb += 32) {
        if (kb + 32 < HSIZE) {
            __builtin_prefetch(xr0 + kb + 32 + klo, 0, 0);
            __builtin_prefetch(xr1 + kb + 32 + klo, 0, 0);
        }
        const f32x4 p0 = __builtin_nontemporal_load((const f32x4*)(xr0 + kb + klo));
        const f32x4 p1 = __builtin_nontemporal_load((const f32x4*)(xr0 + kb + klo + 4));
        const f32x4 p2 = __builtin_nontemporal_load((const f32x4*)(xr0 + kb + khi));
        const f32x4 p3 = __builtin_nontemporal_load((const f32x4*)(xr0 + kb + khi + 4));
        const f32x4 q0 = __builtin_nontemporal_load((const f32x4*)(xr1 + kb + klo));
        const f32x4 q1 = __builtin_nontemporal_load((const f32x4*)(xr1 + kb + klo + 4));
        const f32x4 q2 = __builtin_nontemporal_load((const f32x4*)(xr1 + kb + khi));
        const f32x4 q3 = __builtin_nontemporal_load((const f32x4*)(xr1 + kb + khi + 4));

        const v16bf a0 = pack_a(p0, p1, p2, p3);
        const v16bf a1 = pack_a(q0, q1, q2, q3);

        // B[k][n] = Wg[n][k]: K-run contiguous within a Wg row (32B load).
        const v16bf b0 = *(const v16bf*)(Wgb + (size_t)nb        * HSIZE + kb + kB);
        const v16bf b1 = *(const v16bf*)(Wgb + (size_t)(nb + 16) * HSIZE + kb + kB);

        c00 = __builtin_amdgcn_wmma_f32_16x16x32_bf16(false, a0, false, b0, (short)0, c00, false, false);
        c10 = __builtin_amdgcn_wmma_f32_16x16x32_bf16(false, a1, false, b0, (short)0, c10, false, false);
        c01 = __builtin_amdgcn_wmma_f32_16x16x32_bf16(false, a0, false, b1, (short)0, c01, false, false);
        c11 = __builtin_amdgcn_wmma_f32_16x16x32_bf16(false, a1, false, b1, (short)0, c11, false, false);
    }

    // C layout: lanes 0-15 -> N=lane, M=r; lanes 16-31 -> N=lane-16, M=r+8.
    const int mbase = (lane < 16) ? 0 : 8;
#pragma unroll
    for (int r = 0; r < 8; ++r) {
        const int m0 = row0 + mbase + r;
        const int m1 = m0 + 16;
        gate[(size_t)m0 * EE + nb]      = fast_sigmoid(c00[r]);
        gate[(size_t)m0 * EE + 16 + nb] = fast_sigmoid(c01[r]);
        gate[(size_t)m1 * EE + nb]      = fast_sigmoid(c10[r]);
        gate[(size_t)m1 * EE + 16 + nb] = fast_sigmoid(c11[r]);
    }
}

// ---------------------------------------------------------------------------
// K3: causal depthwise conv (K=4) + gate multiply; emit Y in bf16.
// ---------------------------------------------------------------------------
__global__ void conv_gate_kernel(const float* __restrict__ emb_pre,
                                 const float* __restrict__ conv_w,
                                 const float* __restrict__ conv_b,
                                 const float* __restrict__ gate,
                                 __bf16* __restrict__ Y) {
    const int gid = blockIdx.x * blockDim.x + threadIdx.x;  // NTOK*32 threads
    const int c = gid & 31;
    const int t = gid >> 5;
    const int s = t & (SS - 1);
    float acc = conv_b[c];
#pragma unroll
    for (int j = 0; j < KSZ; ++j) {
        const int sp = s - (KSZ - 1) + j;                   // zero-pad per batch
        if (sp >= 0)
            acc += emb_pre[(size_t)(t - (KSZ - 1) + j) * EE + c] * conv_w[c * KSZ + j];
    }
    Y[(size_t)t * EE + c] = (__bf16)(gate[(size_t)t * EE + c] * acc);
}

// ---------------------------------------------------------------------------
// K4: out = Y @ Wo^T. One wave per 64-row tile: 4 resident A fragments,
// each 32B B load feeds 4 independent WMMAs. out is write-once -> NT stores.
// ---------------------------------------------------------------------------
__global__ void out_wmma_kernel(const __bf16* __restrict__ Y,
                                const __bf16* __restrict__ Wob,
                                float* __restrict__ out) {
    const int lane = threadIdx.x;                // block = exactly 1 wave
    const int row0 = blockIdx.x * 64;

    const int mrow  = lane & 15;
    const int klo   = (lane < 16) ? 0 : 16;
    const int khi   = klo + 8;
    const int kB    = (lane < 16) ? 0 : 16;
    const int nb    = lane & 15;
    const int mbase = (lane < 16) ? 0 : 8;

    // Preload 4 A fragments (held in VGPRs for the whole kernel).
    v16bf a0, a1, a2, a3;
#pragma unroll
    for (int ti = 0; ti < 4; ++ti) {
        const __bf16* yr = Y + (size_t)(row0 + ti * 16 + mrow) * EE;
        const v8bf lo = *(const v8bf*)(yr + klo);
        const v8bf hi = *(const v8bf*)(yr + khi);
        v16bf a;
#pragma unroll
        for (int i = 0; i < 8; ++i) { a[i] = lo[i]; a[8 + i] = hi[i]; }
        if (ti == 0) a0 = a; else if (ti == 1) a1 = a;
        else if (ti == 2) a2 = a; else a3 = a;
    }

    for (int n0 = 0; n0 < HSIZE; n0 += 16) {
        // B[k][n] = Wo[n][k]: contiguous 32B bf16 run within a Wo row.
        const v16bf b = *(const v16bf*)(Wob + (size_t)(n0 + nb) * EE + kB);
        v8f c0 = {}, c1 = {}, c2 = {}, c3 = {};
        c0 = __builtin_amdgcn_wmma_f32_16x16x32_bf16(false, a0, false, b, (short)0, c0, false, false);
        c1 = __builtin_amdgcn_wmma_f32_16x16x32_bf16(false, a1, false, b, (short)0, c1, false, false);
        c2 = __builtin_amdgcn_wmma_f32_16x16x32_bf16(false, a2, false, b, (short)0, c2, false, false);
        c3 = __builtin_amdgcn_wmma_f32_16x16x32_bf16(false, a3, false, b, (short)0, c3, false, false);
#pragma unroll
        for (int r = 0; r < 8; ++r) {
            const size_t col = (size_t)(n0 + nb);
            __builtin_nontemporal_store(c0[r], out + (size_t)(row0      + mbase + r) * HSIZE + col);
            __builtin_nontemporal_store(c1[r], out + (size_t)(row0 + 16 + mbase + r) * HSIZE + col);
            __builtin_nontemporal_store(c2[r], out + (size_t)(row0 + 32 + mbase + r) * HSIZE + col);
            __builtin_nontemporal_store(c3[r], out + (size_t)(row0 + 48 + mbase + r) * HSIZE + col);
        }
    }
}

// ---------------------------------------------------------------------------
// Launch
// ---------------------------------------------------------------------------
extern "C" void kernel_launch(void* const* d_in, const int* in_sizes, int n_in,
                              void* d_out, int out_size, void* d_ws, size_t ws_size,
                              hipStream_t stream) {
    const long long* input_ids = (const long long*)d_in[0];
    const float*     hidden    = (const float*)d_in[1];
    const long long* mapping   = (const long long*)d_in[2];
    const long long* mult      = (const long long*)d_in[3];
    const float*     table     = (const float*)d_in[4];
    const float*     conv_w    = (const float*)d_in[5];
    const float*     conv_b    = (const float*)d_in[6];
    const float*     Wh        = (const float*)d_in[7];
    const float*     Wg        = (const float*)d_in[8];
    const float*     Wo        = (const float*)d_in[9];
    float* out = (float*)d_out;

    // Workspace carve-up (all offsets 256B-aligned).
    char* ws = (char*)d_ws;
    __bf16* Wg_bf   = (__bf16*)(ws + 0);          // 32*2048 bf16   = 128 KB
    __bf16* Wo_bf   = (__bf16*)(ws + 131072);     // 2048*32 bf16   = 128 KB
    float*  emb_pre = (float*)(ws + 262144);      // 16384*32 f32   = 2 MB
    float*  gate    = (float*)(ws + 2359296);     // 16384*32 f32   = 2 MB
    __bf16* Ybf     = (__bf16*)(ws + 4456448);    // 16384*32 bf16  = 1 MB

    cvt_bf16_kernel<<<256, 256, 0, stream>>>(Wg, Wg_bf, EE * HSIZE);
    cvt_bf16_kernel<<<256, 256, 0, stream>>>(Wo, Wo_bf, HSIZE * EE);

    hash_gather_proj_kernel<<<NTOK / 8, 256, 0, stream>>>(
        input_ids, mapping, mult, table, Wh, emb_pre);

    gate_wmma_kernel<<<NTOK / 32, 32, 0, stream>>>(hidden, Wg_bf, gate);

    conv_gate_kernel<<<(NTOK * EE) / 256, 256, 0, stream>>>(
        emb_pre, conv_w, conv_b, gate, Ybf);

    out_wmma_kernel<<<NTOK / 64, 32, 0, stream>>>(Ybf, Wo_bf, out);
}
